// PerceiverResampler_6597069767331
// MI455X (gfx1250) — matrix operand order repaired and checked by reading, not compiled, not run on hardware
//
#include <hip/hip_runtime.h>
#include <hip/hip_bf16.h>
#include <stdint.h>

typedef __attribute__((ext_vector_type(16))) __bf16 v16bf;
typedef __attribute__((ext_vector_type(8)))  __bf16 v8bf;
typedef __attribute__((ext_vector_type(8)))  float  v8f;
typedef __attribute__((ext_vector_type(4)))  int    v4i;

union FragBf { v16bf v; v8bf h[2]; };
union U8bf   { uint4 q; __bf16 e[8]; };

#define WMMA_BF16(a, b, c) \
  __builtin_amdgcn_wmma_f32_16x16x32_bf16(false, (a), false, (b), (short)0, (c), false, false)

// ---------- conversions ----------
__device__ __forceinline__ __bf16 f2bf(float f) {
  union { float f; uint32_t u; } in; in.f = f;
  uint32_t r = in.u + 0x7FFFu + ((in.u >> 16) & 1u);   // round-to-nearest-even
  union { uint16_t s; __bf16 b; } out; out.s = (uint16_t)(r >> 16);
  return out.b;
}
__device__ __forceinline__ uint32_t pack2bf(float a, float b) {
  union { float f; uint32_t u; } x, y; x.f = a; y.f = b;
  uint32_t ra = (x.u + 0x7FFFu + ((x.u >> 16) & 1u)) >> 16;
  uint32_t rb = (y.u + 0x7FFFu + ((y.u >> 16) & 1u)) >> 16;
  return (ra & 0xFFFFu) | (rb << 16);
}
__device__ __forceinline__ uint32_t packbb(__bf16 a, __bf16 b) {
  union { __bf16 b; uint16_t u; } ua, ub; ua.b = a; ub.b = b;
  return (uint32_t)ua.u | ((uint32_t)ub.u << 16);
}

// ---------- CDNA5 async global<->LDS (ASYNCcnt) with sync fallback ----------
#if defined(__has_builtin)
#if __has_builtin(__builtin_amdgcn_global_load_async_to_lds_b128)
#define HAVE_ASYNC_LDS 1
#endif
#endif
#ifndef HAVE_ASYNC_LDS
#define HAVE_ASYNC_LDS 0
#endif

#if HAVE_ASYNC_LDS
typedef __attribute__((address_space(1))) v4i* as1_v4i_ptr;
typedef __attribute__((address_space(3))) v4i* as3_v4i_ptr;
__device__ __forceinline__ void async_copy16(const void* g, void* l) {
  __builtin_amdgcn_global_load_async_to_lds_b128(
      (as1_v4i_ptr)(g), (as3_v4i_ptr)(l), 0, 0);
}
__device__ __forceinline__ void wait_async0() {
#if __has_builtin(__builtin_amdgcn_s_wait_asynccnt)
  __builtin_amdgcn_s_wait_asynccnt(0);
#else
  asm volatile("s_wait_asynccnt 0" ::: "memory");
#endif
}
#if __has_builtin(__builtin_amdgcn_global_store_async_from_lds_b128)
__device__ __forceinline__ void async_store16(void* g, const void* l) {
  __builtin_amdgcn_global_store_async_from_lds_b128(
      (as1_v4i_ptr)(g), (as3_v4i_ptr)(l), 0, 0);
}
#else
__device__ __forceinline__ void async_store16(void* g, const void* l) {
  *(uint4*)g = *(const uint4*)l;
}
#endif
#else
__device__ __forceinline__ void async_copy16(const void* g, void* l) {
  *(uint4*)l = *(const uint4*)g;
}
__device__ __forceinline__ void async_store16(void* g, const void* l) {
  *(uint4*)g = *(const uint4*)l;
}
__device__ __forceinline__ void wait_async0() {}
#endif

// ---------- DPP16 16-lane butterfly reductions (keeps DS pipe free) ----------
template<int CTRL>
__device__ __forceinline__ float dpp_f(float x) {
  return __builtin_bit_cast(float, __builtin_amdgcn_update_dpp(
      0, __builtin_bit_cast(int, x), CTRL, 0xF, 0xF, true));
}
__device__ __forceinline__ float row16_max(float x) {
  x = fmaxf(x, dpp_f<0xB1>(x));    // quad_perm(1,0,3,2)  xor1
  x = fmaxf(x, dpp_f<0x4E>(x));    // quad_perm(2,3,0,1)  xor2
  x = fmaxf(x, dpp_f<0x124>(x));   // row_ror:4
  x = fmaxf(x, dpp_f<0x128>(x));   // row_ror:8
  return x;
}
__device__ __forceinline__ float row16_sum(float x) {
  x += dpp_f<0xB1>(x);
  x += dpp_f<0x4E>(x);
  x += dpp_f<0x124>(x);
  x += dpp_f<0x128>(x);
  return x;
}

// ---------- WMMA fragment loads from LDS ----------
__device__ __forceinline__ v16bf load_frag_a(const __bf16* base, int stride, int lane) {
  int r  = lane & 15;
  int bk = (lane < 16) ? 0 : 8;
  const __bf16* p = base + r * stride + bk;
  FragBf f;
  f.h[0] = *(const v8bf*)(p);
  f.h[1] = *(const v8bf*)(p + 16);
  return f.v;
}
__device__ __forceinline__ v16bf load_frag_b(const __bf16* base, int stride, int lane) {
  int n  = lane & 15;
  int bk = (lane < 16) ? 0 : 8;
  const __bf16* p = base + n * stride + bk;
  FragBf f;
  f.h[0] = *(const v8bf*)(p);
  f.h[1] = *(const v8bf*)(p + 16);
  return f.v;
}

// ---------- prep kernels ----------
__global__ __launch_bounds__(256)
void convert_f32_to_bf16(const float* __restrict__ in, void* __restrict__ out_, long n8) {
  long i = (long)blockIdx.x * 256 + threadIdx.x;
  if (i >= n8) return;
  const float4* g = (const float4*)in + i * 2;
  float4 f0 = g[0], f1 = g[1];
  uint4 o;
  o.x = pack2bf(f0.x, f0.y); o.y = pack2bf(f0.z, f0.w);
  o.z = pack2bf(f1.x, f1.y); o.w = pack2bf(f1.z, f1.w);
  ((uint4*)out_)[i] = o;
}

__global__ __launch_bounds__(256)
void transpose_f32_to_bf16(const float* __restrict__ W, void* __restrict__ WT_, int K, int N) {
  __shared__ __align__(16) __bf16 T[64][72];
  __bf16* WT = (__bf16*)WT_;
  const int k0 = blockIdx.y * 64, n0 = blockIdx.x * 64;
  const int r = threadIdx.x >> 2;
  const int c = (threadIdx.x & 3) * 16;
  const float* g = W + (size_t)(k0 + r) * N + n0 + c;
  float4 f0 = ((const float4*)g)[0], f1 = ((const float4*)g)[1];
  float4 f2 = ((const float4*)g)[2], f3 = ((const float4*)g)[3];
  T[r][c+ 0]=f2bf(f0.x); T[r][c+ 1]=f2bf(f0.y); T[r][c+ 2]=f2bf(f0.z); T[r][c+ 3]=f2bf(f0.w);
  T[r][c+ 4]=f2bf(f1.x); T[r][c+ 5]=f2bf(f1.y); T[r][c+ 6]=f2bf(f1.z); T[r][c+ 7]=f2bf(f1.w);
  T[r][c+ 8]=f2bf(f2.x); T[r][c+ 9]=f2bf(f2.y); T[r][c+10]=f2bf(f2.z); T[r][c+11]=f2bf(f2.w);
  T[r][c+12]=f2bf(f3.x); T[r][c+13]=f2bf(f3.y); T[r][c+14]=f2bf(f3.z); T[r][c+15]=f2bf(f3.w);
  __syncthreads();
  uint4 o0, o1;
  o0.x = packbb(T[c+ 0][r], T[c+ 1][r]); o0.y = packbb(T[c+ 2][r], T[c+ 3][r]);
  o0.z = packbb(T[c+ 4][r], T[c+ 5][r]); o0.w = packbb(T[c+ 6][r], T[c+ 7][r]);
  o1.x = packbb(T[c+ 8][r], T[c+ 9][r]); o1.y = packbb(T[c+10][r], T[c+11][r]);
  o1.z = packbb(T[c+12][r], T[c+13][r]); o1.w = packbb(T[c+14][r], T[c+15][r]);
  uint4* o = (uint4*)(WT + (size_t)(n0 + r) * K + k0 + c);
  o[0] = o0; o[1] = o1;
}

// ---------- GEMM, fp32 inputs converted while staging (small latent GEMMs) ----------
__global__ __launch_bounds__(256)
void gemm_f32_wmma(const float* __restrict__ A, const float* __restrict__ B,
                   void* __restrict__ C_, int M, int N, int K, float scale) {
  __shared__ __align__(16) __bf16 As[64][40];
  __shared__ __align__(16) __bf16 Bs[64][40];   // [n][k]

  const int tid = threadIdx.x, lane = tid & 31, wave = tid >> 5;
  const int m0 = blockIdx.y * 64, n0 = blockIdx.x * 64;
  const int wm = wave & 3, wn = (wave >> 2) * 2;
  v8f acc0 = {}, acc1 = {};

  const int ar = tid >> 2, ac = (tid & 3) * 8;
  const int bk = tid >> 3, bn = (tid & 7) * 8;

  for (int k0 = 0; k0 < K; k0 += 32) {
    {
      const float* g = A + (size_t)(m0 + ar) * K + k0 + ac;
      float4 f0 = *(const float4*)g, f1 = *(const float4*)(g + 4);
      uint4 pk;
      pk.x = pack2bf(f0.x, f0.y); pk.y = pack2bf(f0.z, f0.w);
      pk.z = pack2bf(f1.x, f1.y); pk.w = pack2bf(f1.z, f1.w);
      *(uint4*)&As[ar][ac] = pk;
    }
    {
      const float* g = B + (size_t)(k0 + bk) * N + n0 + bn;
      float4 f0 = *(const float4*)g, f1 = *(const float4*)(g + 4);
      Bs[bn+0][bk]=f2bf(f0.x); Bs[bn+1][bk]=f2bf(f0.y);
      Bs[bn+2][bk]=f2bf(f0.z); Bs[bn+3][bk]=f2bf(f0.w);
      Bs[bn+4][bk]=f2bf(f1.x); Bs[bn+5][bk]=f2bf(f1.y);
      Bs[bn+6][bk]=f2bf(f1.z); Bs[bn+7][bk]=f2bf(f1.w);
    }
    __syncthreads();
    v16bf a  = load_frag_a(&As[wm * 16][0], 40, lane);
    v16bf b0 = load_frag_b(&Bs[wn * 16][0], 40, lane);
    v16bf b1 = load_frag_b(&Bs[(wn + 1) * 16][0], 40, lane);
    acc0 = WMMA_BF16(a, b0, acc0);
    acc1 = WMMA_BF16(a, b1, acc1);
    __syncthreads();
  }

  const int half = lane >> 4, cn = lane & 15;
  for (int t = 0; t < 2; ++t) {
    v8f acc = t ? acc1 : acc0;
    int nglob = n0 + (wn + t) * 16 + cn;
    for (int r = 0; r < 8; ++r) {
      int mglob = m0 + wm * 16 + r + 8 * half;
      ((__bf16*)C_)[(size_t)mglob * N + nglob] = f2bf(acc[r] * scale);
    }
  }
}

// ---------- GEMM, bf16 A + pre-transposed bf16 B ----------
// BM=128, BN=64, BK=32; 8 waves, wave w owns rows [16w,16w+16) x all 4 n-tiles
// (one A-frag feeds 4 WMMAs). Double-buffered async staging, one barrier/step.
template<bool OUT_BF16, bool HAS_BIAS>
__global__ __launch_bounds__(256)
void gemm_bf16_async_wmma(const void* __restrict__ A_, const void* __restrict__ BT_,
                          void* __restrict__ C_, const float* __restrict__ bias,
                          int M, int N, int K, float scale) {
  __shared__ __align__(16) __bf16 As[2][128][40];
  __shared__ __align__(16) __bf16 Bs[2][64][40];   // [n][k]
  __shared__ __align__(16) __bf16 Cs[128][72];     // bf16 epilogue staging
  const __bf16* A  = (const __bf16*)A_;
  const __bf16* BT = (const __bf16*)BT_;

  const int tid = threadIdx.x, lane = tid & 31, wave = tid >> 5;
  const int m0 = blockIdx.y * 128, n0 = blockIdx.x * 64;
  v8f acc[4] = {};

  const int ar = tid >> 2, ac = (tid & 3) * 8;     // A: rows ar, ar+64; B: row ar

  auto stage = [&](int buf, int k0) {
    const __bf16* ga = A + (size_t)(m0 + ar) * K + k0 + ac;
    async_copy16(ga,                  &As[buf][ar][ac]);
    async_copy16(ga + (size_t)64 * K, &As[buf][ar + 64][ac]);
    async_copy16(BT + (size_t)(n0 + ar) * K + k0 + ac, &Bs[buf][ar][ac]);
  };

  const int KT = K >> 5;
  stage(0, 0);
  for (int kt = 0; kt < KT; ++kt) {
    const int cur = kt & 1;
    wait_async0();                 // only the prefetched tile is outstanding
    __syncthreads();               // tile kt visible to all waves
    if (kt + 1 < KT) stage(cur ^ 1, (kt + 1) << 5);   // overlap with compute
    v16bf a = load_frag_a(&As[cur][wave * 16][0], 40, lane);
    #pragma unroll
    for (int t = 0; t < 4; ++t) {
      v16bf b = load_frag_b(&Bs[cur][t * 16][0], 40, lane);
      acc[t] = WMMA_BF16(a, b, acc[t]);
    }
    // frag ds_loads are consumed by the WMMAs (s_wait_dscnt before them), so by
    // the next barrier this wave's reads of buffer cur are done -> WAR-safe.
  }

  const int half = lane >> 4, cn = lane & 15;
  if (OUT_BF16) {
    #pragma unroll
    for (int t = 0; t < 4; ++t)
      #pragma unroll
      for (int r = 0; r < 8; ++r)
        Cs[wave * 16 + r + 8 * half][t * 16 + cn] = f2bf(acc[t][r] * scale);
    __syncthreads();
    const int row = tid >> 1, ch = (tid & 1) * 32;
    __bf16* g = (__bf16*)C_ + (size_t)(m0 + row) * N + n0 + ch;
    async_store16(g +  0, &Cs[row][ch +  0]);
    async_store16(g +  8, &Cs[row][ch +  8]);
    async_store16(g + 16, &Cs[row][ch + 16]);
    async_store16(g + 24, &Cs[row][ch + 24]);
    wait_async0();
  } else {
    #pragma unroll
    for (int t = 0; t < 4; ++t) {
      int nglob = n0 + t * 16 + cn;
      float bv = HAS_BIAS ? bias[nglob] : 0.0f;
      #pragma unroll
      for (int r = 0; r < 8; ++r) {
        int mglob = m0 + wave * 16 + r + 8 * half;
        ((float*)C_)[(size_t)mglob * N + nglob] = acc[t][r] * scale + bv;
      }
    }
  }
}

// ---------- flash cross-attention ----------
__global__ __launch_bounds__(128)
void attn_flash_wmma(const void* __restrict__ qb_,   // bf16 [64][1024], scale pre-applied
                     const void* __restrict__ kvb_,  // bf16 [32768][2048]  (k | v)
                     const void* __restrict__ lkvb_, // bf16 [64][2048]     (lk | lv)
                     void* __restrict__ outb_)       // bf16 [2048][1024]
{
  const __bf16* qb   = (const __bf16*)qb_;
  const __bf16* kvb  = (const __bf16*)kvb_;
  const __bf16* lkvb = (const __bf16*)lkvb_;
  __bf16* outb = (__bf16*)outb_;

  __shared__ __align__(16) __bf16 Qs[64][72];      // [i][dh]; reused as out staging
  __shared__ __align__(16) __bf16 Ks[32][72];      // [j][dh]
  __shared__ __align__(16) __bf16 Vs[64][40];      // [dh][j]
  __shared__ __align__(16) __bf16 Ps[4][16][40];   // per-wave P re-layout

  const int tid = threadIdx.x, lane = tid & 31, wave = tid >> 5;
  const int half = lane >> 4;
  const int bid = blockIdx.x;
  const int b = bid >> 6, h = (bid >> 2) & 15, m = bid & 3;
  const int bm = b * 4 + m;

  { // stage Q (64x64 head slice): 4x16B async per thread
    int i = tid >> 1, c = (tid & 1) * 32;
    const __bf16* g = qb + (size_t)i * 1024 + h * 64 + c;
    async_copy16(g +  0, &Qs[i][c +  0]);
    async_copy16(g +  8, &Qs[i][c +  8]);
    async_copy16(g + 16, &Qs[i][c + 16]);
    async_copy16(g + 24, &Qs[i][c + 24]);
    wait_async0();
  }
  __syncthreads();

  v16bf qf0 = load_frag_a(&Qs[wave * 16][0],  72, lane);
  v16bf qf1 = load_frag_a(&Qs[wave * 16][32], 72, lane);

  v8f O0 = {}, O1 = {}, O2 = {}, O3 = {};
  float row_m[8], row_l[8];
  #pragma unroll
  for (int r = 0; r < 8; ++r) { row_m[r] = -3.0e38f; row_l[r] = 0.0f; }

  const int jl = tid >> 2;           // local key row 0..31
  const int cb = (tid & 3) * 16;     // dh base

  for (int j0 = 0; j0 < 1088; j0 += 32) {
    int j = j0 + jl;
    const __bf16* row = (j < 1024) ? (kvb + (size_t)(bm * 1024 + j) * 2048)
                                   : (lkvb + (size_t)(j - 1024) * 2048);
    const __bf16* gk = row + h * 64 + cb;
    const __bf16* gv = row + 1024 + h * 64 + cb;
    async_copy16(gk + 0, &Ks[jl][cb + 0]);
    async_copy16(gk + 8, &Ks[jl][cb + 8]);
    U8bf v0, v1;
    v0.q = *(const uint4*)(gv);
    v1.q = *(const uint4*)(gv + 8);
    #pragma unroll
    for (int e = 0; e < 8; ++e) {
      Vs[cb + e][jl]     = v0.e[e];
      Vs[cb + 8 + e][jl] = v1.e[e];
    }
    wait_async0();
    __syncthreads();

    v8f s0 = {}, s1 = {};
    { v16bf kb = load_frag_b(&Ks[0][0],   72, lane); s0 = WMMA_BF16(qf0, kb, s0); }
    { v16bf kb = load_frag_b(&Ks[0][32],  72, lane); s0 = WMMA_BF16(qf1, kb, s0); }
    { v16bf kb = load_frag_b(&Ks[16][0],  72, lane); s1 = WMMA_BF16(qf0, kb, s1); }
    { v16bf kb = load_frag_b(&Ks[16][32], 72, lane); s1 = WMMA_BF16(qf1, kb, s1); }

    #pragma unroll
    for (int r = 0; r < 8; ++r) {
      float cm = row16_max(fmaxf(s0[r], s1[r]));
      float nm = fmaxf(row_m[r], cm);
      float alpha = __expf(row_m[r] - nm);
      row_m[r] = nm;
      float p0 = __expf(s0[r] - nm);
      float p1 = __expf(s1[r] - nm);
      float ps = row16_sum(p0 + p1);
      row_l[r] = row_l[r] * alpha + ps;
      O0[r] *= alpha; O1[r] *= alpha; O2[r] *= alpha; O3[r] *= alpha;
      int rr = r + 8 * half;
      Ps[wave][rr][(lane & 15)]      = f2bf(p0);
      Ps[wave][rr][16 + (lane & 15)] = f2bf(p1);
    }
    asm volatile("s_wait_dscnt 0" ::: "memory");   // P store -> reload (same wave)

    v16bf pa = load_frag_a(&Ps[wave][0][0], 40, lane);
    { v16bf vb = load_frag_b(&Vs[0][0],  40, lane); O0 = WMMA_BF16(pa, vb, O0); }
    { v16bf vb = load_frag_b(&Vs[16][0], 40, lane); O1 = WMMA_BF16(pa, vb, O1); }
    { v16bf vb = load_frag_b(&Vs[32][0], 40, lane); O2 = WMMA_BF16(pa, vb, O2); }
    { v16bf vb = load_frag_b(&Vs[48][0], 40, lane); O3 = WMMA_BF16(pa, vb, O3); }
    __syncthreads();
  }

  // normalize into LDS (reuse Qs), then coalesced async 16B stores
  #pragma unroll
  for (int r = 0; r < 8; ++r) {
    int rowi = wave * 16 + r + 8 * half;
    float inv = 1.0f / row_l[r];
    Qs[rowi][(lane & 15) +  0] = f2bf(O0[r] * inv);
    Qs[rowi][(lane & 15) + 16] = f2bf(O1[r] * inv);
    Qs[rowi][(lane & 15) + 32] = f2bf(O2[r] * inv);
    Qs[rowi][(lane & 15) + 48] = f2bf(O3[r] * inv);
  }
  __syncthreads();
  {
    const int row = tid >> 1, ch = (tid & 1) * 32;
    __bf16* g = outb + ((size_t)bm * 64 + row) * 1024 + h * 64 + ch;
    async_store16(g +  0, &Qs[row][ch +  0]);
    async_store16(g +  8, &Qs[row][ch +  8]);
    async_store16(g + 16, &Qs[row][ch + 16]);
    async_store16(g + 24, &Qs[row][ch + 24]);
    wait_async0();
  }
}

// ---------- host launcher ----------
extern "C" void kernel_launch(void* const* d_in, const int* in_sizes, int n_in,
                              void* d_out, int out_size, void* d_ws, size_t ws_size,
                              hipStream_t stream) {
  const float* x    = (const float*)d_in[0];  // [8,4,1024,1024]
  const float* lat  = (const float*)d_in[1];  // [64,1024]
  const float* Wq   = (const float*)d_in[2];  // [1024,1024]
  const float* Wkv  = (const float*)d_in[3];  // [1024,2048]
  const float* Wout = (const float*)d_in[4];  // [1024,1024]
  const float* bout = (const float*)d_in[5];  // [1024]
  float* out = (float*)d_out;                 // [8,4,64,1024] fp32

  char* ws = (char*)d_ws;
  size_t off = 0;
  void* q_buf   = ws + off; off += (size_t)64 * 1024 * 2;        // q bf16
  void* lkv_buf = ws + off; off += (size_t)64 * 2048 * 2;        // lkv bf16
  void* kv_buf  = ws + off; off += (size_t)32768 * 2048 * 2;     // kv bf16
  void* at_buf  = ws + off; off += (size_t)2048 * 1024 * 2;      // attn bf16
  void* xb_buf  = ws + off; off += (size_t)32768 * 1024 * 2;     // x bf16
  void* wkvT    = ws + off; off += (size_t)2048 * 1024 * 2;      // Wkv^T bf16
  void* woutT   = ws + off; off += (size_t)1024 * 1024 * 2;      // Wout^T bf16

  dim3 blk(256);
  // prep: x -> bf16 (67MB; fits in 192MB L2 for the kv GEMM's 32 re-reads)
  convert_f32_to_bf16<<<dim3(16384), blk, 0, stream>>>(x, xb_buf, (long)32768 * 1024 / 8);
  // prep: Wkv^T, Wout^T -> bf16 [N][K]
  transpose_f32_to_bf16<<<dim3(32, 16), blk, 0, stream>>>(Wkv, wkvT, 1024, 2048);
  transpose_f32_to_bf16<<<dim3(16, 16), blk, 0, stream>>>(Wout, woutT, 1024, 1024);
  // q = (latents @ Wq) * dh^-0.5 ; lkv = latents @ Wkv
  gemm_f32_wmma<<<dim3(16, 1), blk, 0, stream>>>(lat, Wq,  q_buf,   64, 1024, 1024, 0.125f);
  gemm_f32_wmma<<<dim3(32, 1), blk, 0, stream>>>(lat, Wkv, lkv_buf, 64, 2048, 1024, 1.0f);
  // kv = x @ Wkv  (dominant GEMM: BM=128 tile, double-buffered async staging)
  gemm_bf16_async_wmma<true, false><<<dim3(32, 256), blk, 0, stream>>>(
      xb_buf, wkvT, kv_buf, nullptr, 32768, 2048, 1024, 1.0f);
  // flash attention per (b,h,m)
  attn_flash_wmma<<<dim3(512), dim3(128), 0, stream>>>(q_buf, kv_buf, lkv_buf, at_buf);
  // out = attn @ Wout + bout
  gemm_bf16_async_wmma<false, true><<<dim3(16, 16), blk, 0, stream>>>(
      at_buf, woutT, out, bout, 2048, 1024, 1024, 1.0f);
}
